// MaximalCodingRateReduction_61375082660446
// MI455X (gfx1250) — matrix operand order repaired and checked by reading, not compiled, not run on hardware
//
#include <hip/hip_runtime.h>
#include <hip/hip_bf16.h>

#define P     128
#define NCLS  10
#define NS    64          // slices per class -> 640 workgroups
#define BATCH 256         // compacted rows staged in LDS per inner pass
#define XSTR  (BATCH + 8) // padded col-major stride (528B, 16B aligned)

#define GAM1f 1.0f
#define GAM2f 1.0f
#define EPSf  0.01f

typedef __attribute__((ext_vector_type(16))) __bf16 v16bf;
typedef __attribute__((ext_vector_type(8)))  float  v8f;

union FragBf { uint4 u4[2]; v16bf v; };

__device__ __forceinline__ unsigned short f2bf(float f) {
  unsigned u = __float_as_uint(f);
  unsigned r = u + 0x7FFFu + ((u >> 16) & 1u);   // round-to-nearest-even
  return (unsigned short)(r >> 16);
}

// ---------------------------------------------------------------------------
// Zero G[10][128][128] + counts + logdets in d_ws (must run every call).
// ---------------------------------------------------------------------------
__global__ void mcr_zero(float* w, int n) {
  int i = blockIdx.x * blockDim.x + threadIdx.x;
  if (i < n) w[i] = 0.0f;
}

// ---------------------------------------------------------------------------
// One K-step (32 rows) for tile row RW: branch-free, fully unrolled so the
// scheduler can pipeline ds_load_b128 against v_wmma (as in round 1).
// ---------------------------------------------------------------------------
template <int RW>
__device__ __forceinline__ void kstep(const unsigned short* __restrict__ Xcm,
                                      const unsigned short* __restrict__ Xrm,
                                      int k0, int l16, int hlf, int lane,
                                      v8f (&acc)[8]) {
  // A = X^T tile (16 cols of x, 32 K-rows), ISA 16-bit A 16x32 layout:
  // lane<16: elems 0..7 = K k0..k0+7, 8..15 = K k0+16..23 ; lane>=16: +8
  FragBf A;
  const unsigned short* cp = Xcm + (RW * 16 + l16) * XSTR + k0 + hlf * 8;
  A.u4[0] = *(const uint4*)(cp);
  A.u4[1] = *(const uint4*)(cp + 16);
  #pragma unroll
  for (int j = RW; j < 8; ++j) {
    // B 32x16: lane = N col (l16), elems 0..7 = K k0+8h.., 8..15 = K k0+16+8h..
    FragBf B;
    const unsigned short* rp = Xcm + (j * 16 + l16) * XSTR + k0 + hlf * 8;
    B.u4[0] = *(const uint4*)(rp);
    B.u4[1] = *(const uint4*)(rp + 16);
    acc[j] = __builtin_amdgcn_wmma_f32_16x16x32_bf16(
        false, A.v, false, B.v, (short)0, acc[j], false, false);
  }
}

// ---------------------------------------------------------------------------
// Per-class Gram via WMMA, exploiting symmetry (upper-triangle tiles only).
// Grid = NCLS*NS blocks of 256 threads (8 waves).
// Block (c,s): scan y over slice s, gather class-c rows, compact into LDS
// (row-major + col-major bf16), K-loop of v_wmma_f32_16x16x32_bf16 over the
// wave's tile row, then one atomic flush (with transpose mirror).
// Wave w owns tile row rw = (w<4 ? w : 11-w) and tiles j = rw..7; the pairing
// (0,7),(1,6),(2,5),(3,4) gives each SIMD exactly 9 tiles per K-step.
// The tile row is readfirstlane'd and dispatched through a uniform switch so
// no EXEC masking ever surrounds the WMMAs.
// ---------------------------------------------------------------------------
__global__ __launch_bounds__(256) void mcr_gram(
    const float* __restrict__ x, const int* __restrict__ y,
    float* __restrict__ G, int* __restrict__ counts, int m, int RS)
{
  extern __shared__ char smem[];
  int*            sIdx = (int*)smem;                              // RS ints
  unsigned short* Xrm  = (unsigned short*)(smem + (size_t)RS * 4);// BATCH x P
  unsigned short* Xcm  = Xrm + BATCH * P;                         // P x XSTR
  int*            sCnt = (int*)(Xcm + P * XSTR);

  const int tid = threadIdx.x;
  const int cls = blockIdx.x / NS;
  const int slc = blockIdx.x % NS;
  const int r0  = slc * RS;
  const int r1  = (r0 + RS < m) ? (r0 + RS) : m;

  if (tid == 0) sCnt[0] = 0;
  __syncthreads();
  for (int r = r0 + tid; r < r1; r += blockDim.x)
    if (y[r] == cls) { int p = atomicAdd(sCnt, 1); sIdx[p] = r; }
  __syncthreads();
  const int nC = sCnt[0];

  v8f acc[8];
  #pragma unroll
  for (int j = 0; j < 8; ++j) acc[j] = (v8f){0,0,0,0,0,0,0,0};

  const int lane = tid & 31;
  const int wv   = tid >> 5;
  // tile row, SIMD-balanced; readfirstlane -> provably wave-uniform scalar
  const int rws  = __builtin_amdgcn_readfirstlane((wv < 4) ? wv : 11 - wv);
  const int hlf  = lane >> 4;                 // A-layout lane half
  const int l16  = lane & 15;

  const int nBatch = (nC + BATCH - 1) / BATCH;
  for (int b = 0; b < nBatch; ++b) {
    __syncthreads();                               // LDS reuse fence
    const int rows   = (nC - b * BATCH < BATCH) ? (nC - b * BATCH) : BATCH;
    const int padded = (rows + 31) & ~31;          // zero-pad to K-step
    const int tin    = tid & 7;                    // 8 threads per row
    for (int s0 = tid >> 3; s0 < padded; s0 += 32) {
      if (s0 < rows) {
        const int g = sIdx[b * BATCH + s0];
        const float4* src = (const float4*)(x + (size_t)g * P);
        #pragma unroll
        for (int q = 0; q < 4; ++q) {
          float4 v = src[q * 8 + tin];             // coalesced 16B/lane
          int cb = (q * 8 + tin) * 4;
          ushort4 pk;
          pk.x = f2bf(v.x); pk.y = f2bf(v.y); pk.z = f2bf(v.z); pk.w = f2bf(v.w);
          *(ushort4*)(Xrm + s0 * P + cb) = pk;
          Xcm[(cb + 0) * XSTR + s0] = pk.x;
          Xcm[(cb + 1) * XSTR + s0] = pk.y;
          Xcm[(cb + 2) * XSTR + s0] = pk.z;
          Xcm[(cb + 3) * XSTR + s0] = pk.w;
        }
      } else {                                     // zero padding rows
        #pragma unroll
        for (int q = 0; q < 4; ++q) {
          int cb = (q * 8 + tin) * 4;
          *(ushort4*)(Xrm + s0 * P + cb) = make_ushort4(0, 0, 0, 0);
          Xcm[(cb + 0) * XSTR + s0] = 0;
          Xcm[(cb + 1) * XSTR + s0] = 0;
          Xcm[(cb + 2) * XSTR + s0] = 0;
          Xcm[(cb + 3) * XSTR + s0] = 0;
        }
      }
    }
    __syncthreads();

    const int steps = padded >> 5;
    // Uniform scalar dispatch: each case is a branch-free K-loop.
    switch (rws) {
      case 0: for (int st = 0; st < steps; ++st) kstep<0>(Xcm, Xrm, st << 5, l16, hlf, lane, acc); break;
      case 1: for (int st = 0; st < steps; ++st) kstep<1>(Xcm, Xrm, st << 5, l16, hlf, lane, acc); break;
      case 2: for (int st = 0; st < steps; ++st) kstep<2>(Xcm, Xrm, st << 5, l16, hlf, lane, acc); break;
      case 3: for (int st = 0; st < steps; ++st) kstep<3>(Xcm, Xrm, st << 5, l16, hlf, lane, acc); break;
      case 4: for (int st = 0; st < steps; ++st) kstep<4>(Xcm, Xrm, st << 5, l16, hlf, lane, acc); break;
      case 5: for (int st = 0; st < steps; ++st) kstep<5>(Xcm, Xrm, st << 5, l16, hlf, lane, acc); break;
      case 6: for (int st = 0; st < steps; ++st) kstep<6>(Xcm, Xrm, st << 5, l16, hlf, lane, acc); break;
      default:for (int st = 0; st < steps; ++st) kstep<7>(Xcm, Xrm, st << 5, l16, hlf, lane, acc); break;
    }
  }

  if (nC > 0) {
    float* Gc = G + (size_t)cls * P * P;
    const int moff = rws * 16 + (hlf << 3);  // C layout: lane>=16 -> M+8
    #pragma unroll
    for (int j = 0; j < 8; ++j) {
      if (j >= rws) {                        // uniform scalar guard
        #pragma unroll
        for (int r = 0; r < 8; ++r) {
          const int M = moff + r;
          const int N = j * 16 + l16;
          const float v = acc[j][r];
          unsafeAtomicAdd(&Gc[(size_t)M * P + N], v);
          if (j > rws)                       // mirror strictly-upper tiles
            unsafeAtomicAdd(&Gc[(size_t)N * P + M], v);
        }
      }
    }
    if (tid == 0) atomicAdd(&counts[cls], nC);
  }
}

// ---------------------------------------------------------------------------
// logdet(I + s*G) for 12 SPD 128x128 matrices (b=0: GAM1*scalar*Gtot,
// b=1: scalar*Gtot, b>=2: class c=b-2). Gaussian elimination in LDS.
// ---------------------------------------------------------------------------
__global__ void mcr_logdet(const float* __restrict__ G,
                           const int* __restrict__ counts,
                           float* __restrict__ ld, int m)
{
  extern __shared__ char smem[];
  float* A = (float*)smem;                 // 128 x 129
  const int b = blockIdx.x;
  const int r = threadIdx.x;

  float s;
  const float* src = nullptr;
  if (b < 2) {
    s = (b == 0 ? GAM1f : 1.0f) * ((float)P / ((float)m * EPSf));
  } else {
    int cnt = counts[b - 2];
    float sc = cnt > 0 ? (float)cnt : 1.0f;
    s = (float)P / (sc * EPSf);
    src = G + (size_t)(b - 2) * P * P;
  }
  for (int c = 0; c < P; ++c) {
    float g;
    if (b < 2) {
      g = 0.0f;
      for (int cc = 0; cc < NCLS; ++cc) g += G[(size_t)cc * P * P + r * P + c];
    } else {
      g = src[r * P + c];
    }
    A[r * 129 + c] = (r == c ? 1.0f : 0.0f) + s * g;
  }
  __syncthreads();

  float logsum = 0.0f;
  for (int k = 0; k < P; ++k) {
    float akk = A[k * 129 + k];
    if (r == 0) logsum += logf(akk);
    if (r > k) {
      float f = A[r * 129 + k] / akk;
      for (int c = k + 1; c < P; ++c) A[r * 129 + c] -= f * A[k * 129 + c];
    }
    __syncthreads();
  }
  if (r == 0) ld[b] = logsum;
}

// ---------------------------------------------------------------------------
// Scalar combine -> 4 outputs.
// ---------------------------------------------------------------------------
__global__ void mcr_finalize(const float* __restrict__ ld,
                             const int* __restrict__ counts,
                             float* __restrict__ out, int m)
{
  float de = 0.5f * ld[0];
  float dt = 0.5f * ld[1];
  float comp = 0.0f;
  for (int c = 0; c < NCLS; ++c)
    if (counts[c] > 0) comp += ld[2 + c] * (float)counts[c];
  comp = comp / (2.0f * (float)m);
  out[0] = GAM2f * (-de) + comp;
  out[1] = de;
  out[2] = dt;
  out[3] = comp;
}

// ---------------------------------------------------------------------------
extern "C" void kernel_launch(void* const* d_in, const int* in_sizes, int n_in,
                              void* d_out, int out_size, void* d_ws, size_t ws_size,
                              hipStream_t stream) {
  const float* x = (const float*)d_in[0];
  const int*   y = (const int*)d_in[1];
  const int    m = in_sizes[0] / P;

  float* G      = (float*)d_ws;                                 // 10*128*128 f32
  int*   counts = (int*)((char*)d_ws + (size_t)NCLS * P * P * 4);
  float* ld     = (float*)((char*)d_ws + (size_t)NCLS * P * P * 4 + 64);
  float* out    = (float*)d_out;

  int RS = (m + NS - 1) / NS;
  RS = (RS + 31) & ~31;

  size_t sh_gram = (size_t)RS * 4 + (size_t)BATCH * P * 2 +
                   (size_t)P * XSTR * 2 + 16;                   // ~146 KB
  size_t sh_ld   = (size_t)128 * 129 * 4;                       // ~66 KB
  hipFuncSetAttribute((const void*)mcr_gram,
                      hipFuncAttributeMaxDynamicSharedMemorySize, (int)sh_gram);
  hipFuncSetAttribute((const void*)mcr_logdet,
                      hipFuncAttributeMaxDynamicSharedMemorySize, (int)sh_ld);

  const int nz = NCLS * P * P + 32;                             // G + counts + ld
  mcr_zero<<<(nz + 255) / 256, 256, 0, stream>>>((float*)d_ws, nz);
  mcr_gram<<<NCLS * NS, 256, sh_gram, stream>>>(x, y, G, counts, m, RS);
  mcr_logdet<<<NCLS + 2, 128, sh_ld, stream>>>(G, counts, ld, m);
  mcr_finalize<<<1, 1, 0, stream>>>(ld, counts, out, m);
}
